// LinearBin_27994596835418
// MI455X (gfx1250) — matrix operand order repaired and checked by reading, not compiled, not run on hardware
//
#include <hip/hip_runtime.h>

typedef __attribute__((ext_vector_type(4)))  _Float16 v4h;
typedef __attribute__((ext_vector_type(8)))  _Float16 v8h;
typedef __attribute__((ext_vector_type(16))) _Float16 v16h;
typedef __attribute__((ext_vector_type(4)))  float    v4f;
typedef __attribute__((ext_vector_type(8)))  float    v8f;

#define IN_F        512
#define OUT_F       512
#define ROWS_PER_WG 64
#define NBATCH      131072

// ---------------------------------------------------------------------------
// Stage 1: binarize weight (f32 -> f16 {-1,+1}), keep [OUT][IN] row-major so
// WMMA B fragments are contiguous 16B chunks along K.
// 512*512 = 262144 elems = 65536 v4f; 256 blocks x 256 threads, 1 v4f each.
// ---------------------------------------------------------------------------
__global__ __launch_bounds__(256) void binarize_w_kernel(
    const float* __restrict__ w, _Float16* __restrict__ wb) {
  int i = blockIdx.x * 256 + threadIdx.x;          // v4f index
  v4f v = ((const v4f*)w)[i];
  v4h o;
#pragma unroll
  for (int j = 0; j < 4; ++j)
    o[j] = (v[j] >= 0.0f) ? (_Float16)1.0f : (_Float16)-1.0f;
  ((v4h*)wb)[i] = o;
}

// ---------------------------------------------------------------------------
// Stage 2: out[64rows x 512] per workgroup.
// 8 waves: wave w owns N columns [w*64, w*64+64), covers all 4 M-tiles (16 rows
// each) so every B fragment is reused 4x (4 independent WMMA accumulation
// chains -> good XDL pipelining).
// ---------------------------------------------------------------------------
__global__ __launch_bounds__(256) void linbin_gemm_kernel(
    const float* __restrict__ x, const _Float16* __restrict__ wb,
    const float* __restrict__ bias, float* __restrict__ out) {
  __shared__ _Float16 lds_x[ROWS_PER_WG * IN_F];   // 64 KB

  const int  tid   = threadIdx.x;
  const long mbase = (long)blockIdx.x * ROWS_PER_WG;

  // ---- load x tile, convert f32->f16 into LDS (coalesced, streaming) ----
  {
    const v4f* xp = (const v4f*)(x + mbase * IN_F);
#pragma unroll
    for (int i = 0; i < 32; ++i) {                 // 8192 v4f / 256 threads
      int idx = tid + i * 256;
      v4f v = __builtin_nontemporal_load(&xp[idx]);
      v4h h;
#pragma unroll
      for (int j = 0; j < 4; ++j) h[j] = (_Float16)v[j];
      *(v4h*)(&lds_x[idx * 4]) = h;                // ds_store_b64
    }
  }
  __syncthreads();

  const int wave  = tid >> 5;                      // 0..7
  const int lane  = tid & 31;
  const int lhalf = lane >> 4;                     // 0 or 1 (half-wave)
  const int l15   = lane & 15;

  for (int nt = 0; nt < 4; ++nt) {
    const int ncol = wave * 64 + nt * 16 + l15;    // this lane's output column

    // C/D layout: lane covers col (lane&15), rows r + 8*(lane>>4); bias is
    // per-column -> broadcast into all 8 accumulator registers.
    const float bv = bias[ncol];
    v8f acc[4];
#pragma unroll
    for (int mt = 0; mt < 4; ++mt)
#pragma unroll
      for (int r = 0; r < 8; ++r) acc[mt][r] = bv;

    // B fragment base: lane holds K = k0 + (lane>>4)*16 .. +15 of row `ncol`.
    const _Float16* wrow = wb + (long)ncol * IN_F + lhalf * 16;

    for (int k0 = 0; k0 < IN_F; k0 += 32) {
      // ---- B fragment: 2x global_load_b128 (hits L2/L0, wb is 512 KB) ----
      v8h blo = *(const v8h*)(wrow + k0);
      v8h bhi = *(const v8h*)(wrow + k0 + 8);
      v16h bf;
#pragma unroll
      for (int i = 0; i < 8; ++i) { bf[i] = blo[i]; bf[i + 8] = bhi[i]; }

#pragma unroll
      for (int mt = 0; mt < 4; ++mt) {
        // A fragment: lane holds row (lane&15), K = k0 + (lane>>4)*16 .. +15
        // contiguous in halves 0..15. Two ds_load_b128.
        const _Float16* ap = &lds_x[(mt * 16 + l15) * IN_F + k0 + lhalf * 16];
        v8h alo = *(const v8h*)ap;
        v8h ahi = *(const v8h*)(ap + 8);
        v16h af;
#pragma unroll
        for (int i = 0; i < 8; ++i) { af[i] = alo[i]; af[i + 8] = ahi[i]; }

        acc[mt] = __builtin_amdgcn_wmma_f32_16x16x32_f16(
            /*neg_a=*/false, af, /*neg_b=*/false, bf,
            /*c_mod=*/(short)0, acc[mt], /*reuse_a=*/false, /*reuse_b=*/false);
      }
    }

    // ---- store D: lane writes col ncol, rows mbase + mt*16 + 8*lhalf + r ----
#pragma unroll
    for (int mt = 0; mt < 4; ++mt) {
      float* op = out + (mbase + mt * 16 + lhalf * 8) * OUT_F + ncol;
#pragma unroll
      for (int r = 0; r < 8; ++r)
        __builtin_nontemporal_store(acc[mt][r], op + (long)r * OUT_F);
    }
  }
}

extern "C" void kernel_launch(void* const* d_in, const int* in_sizes, int n_in,
                              void* d_out, int out_size, void* d_ws, size_t ws_size,
                              hipStream_t stream) {
  (void)in_sizes; (void)n_in; (void)out_size; (void)ws_size;
  const float* x      = (const float*)d_in[0];   // [131072, 512] f32
  const float* weight = (const float*)d_in[1];   // [512, 512]    f32
  const float* bias   = (const float*)d_in[2];   // [512]         f32
  float*       out    = (float*)d_out;           // [131072, 512] f32
  _Float16*    wb     = (_Float16*)d_ws;         // 512 KB binarized weight

  // Stage 1: binarize weights into workspace (f16, row-major [OUT][IN]).
  binarize_w_kernel<<<(OUT_F * IN_F / 4) / 256, 256, 0, stream>>>(weight, wb);

  // Stage 2: WMMA GEMM. 131072 / 64 = 2048 workgroups.
  linbin_gemm_kernel<<<NBATCH / ROWS_PER_WG, 256, 0, stream>>>(x, wb, bias, out);
}
